// WeightedMSELoss_79731772883678
// MI455X (gfx1250) — compile-verified, matrix operand order checked by reading, not run on hardware
//
#include <hip/hip_runtime.h>
#include <math.h>

typedef int   v8i  __attribute__((ext_vector_type(8)));
typedef float f32x4 __attribute__((ext_vector_type(4)));

#define SLICE_ELEMS (512 * 512)            // 262144 elements per (b,c)
#define NBC         208                    // 16 * 13 slices
#define G_PER_SLICE 8                      // workgroups cooperating on one slice
#define THREADS     256                    // 8 wave32s
#define F4_PER_SLICE (SLICE_ELEMS / 4)     // 65536 float4 per slice
#define F4_PER_PASS  (G_PER_SLICE * THREADS) // 2048 float4 per pass over slice
#define OUTER        (F4_PER_SLICE / F4_PER_PASS / 8) // 4 WMMA iterations/thread

// ---------------------------------------------------------------------------
// Kernel 1: streaming masked reduction.
//   err  : f32 VALU FMA accumulation (exact)
//   count: packed 0/1 mask bytes -> V_WMMA_I32_16X16X64_IU8 with all-ones B
//          (1024 mask bytes accumulated exactly per instruction)
// Each workgroup writes ONE (err, count) partial -> no atomics, deterministic.
// ---------------------------------------------------------------------------
__global__ __launch_bounds__(THREADS) void wmse_partial_kernel(
    const float* __restrict__ outs, const float* __restrict__ tgts,
    float* __restrict__ partial) {
  const int tid   = threadIdx.x;
  const int slice = blockIdx.x / G_PER_SLICE;
  const int g     = blockIdx.x % G_PER_SLICE;
  const int flat  = g * THREADS + tid;                 // 0 .. 2047
  const size_t base_f4 = (size_t)slice * F4_PER_SLICE;

  const f32x4* O4 = (const f32x4*)outs;
  const f32x4* T4 = (const f32x4*)tgts;

  float err = 0.0f;
  v8i acc = {0, 0, 0, 0, 0, 0, 0, 0};   // i32 WMMA accumulator (count)
  v8i bones;                             // B matrix: every i8 element == 1
#pragma unroll
  for (int i = 0; i < 8; ++i) bones[i] = 0x01010101;

#pragma unroll 1
  for (int w = 0; w < OUTER; ++w) {
    v8i a;                               // 32 mask bytes (this lane's A slice)
#pragma unroll
    for (int u = 0; u < 8; ++u) {
      const size_t idx = base_f4 + (size_t)((w * 8 + u) * F4_PER_PASS + flat);
      f32x4 o = __builtin_nontemporal_load(&O4[idx]);  // streaming: TH=NT,
      f32x4 t = __builtin_nontemporal_load(&T4[idx]);  // don't thrash 192MB L2
      const int m0 = t[0] > 0.0f;
      const int m1 = t[1] > 0.0f;
      const int m2 = t[2] > 0.0f;
      const int m3 = t[3] > 0.0f;
      const float d0 = m0 ? (o[0] - t[0]) : 0.0f;
      const float d1 = m1 ? (o[1] - t[1]) : 0.0f;
      const float d2 = m2 ? (o[2] - t[2]) : 0.0f;
      const float d3 = m3 ? (o[3] - t[3]) : 0.0f;
      err = fmaf(d0, d0, err);
      err = fmaf(d1, d1, err);
      err = fmaf(d2, d2, err);
      err = fmaf(d3, d3, err);
      a[u] = m0 | (m1 << 8) | (m2 << 16) | (m3 << 24);  // 4 mask bytes/dword
    }
    // D += A(16x64 i8 masks) x B(64x16 i8 ones): exact integer mask count.
    acc = __builtin_amdgcn_wmma_i32_16x16x64_iu8(
        /*sgn_a=*/false, a, /*sgn_b=*/false, bones, acc,
        /*reuse_a=*/false, /*reuse_b=*/false);
  }

  // With B==ones every D column is identical: lanes 0-15 hold sum of rows 0-7,
  // lanes 16-31 hold sum of rows 8-15 -> total = s + swap16(s).
  int s = 0;
#pragma unroll
  for (int i = 0; i < 8; ++i) s += acc[i];
  int cnt = s + __shfl_xor(s, 16, 32);

  // wave32 tree-reduce the f32 error
#pragma unroll
  for (int off = 16; off > 0; off >>= 1) err += __shfl_xor(err, off, 32);

  __shared__ float serr[THREADS / 32];
  __shared__ int   scnt[THREADS / 32];
  const int wid = tid >> 5;
  if ((tid & 31) == 0) { serr[wid] = err; scnt[wid] = cnt; }
  __syncthreads();
  if (tid == 0) {
    float e = 0.0f;
    int   c = 0;
#pragma unroll
    for (int i = 0; i < THREADS / 32; ++i) { e += serr[i]; c += scnt[i]; }
    partial[2 * blockIdx.x + 0] = e;
    partial[2 * blockIdx.x + 1] = (float)c;   // counts <= 262144, exact in f32
  }
}

// ---------------------------------------------------------------------------
// Kernel 2: 208-slice epilogue + weighted mean -> scalar.
// ---------------------------------------------------------------------------
__global__ __launch_bounds__(256) void wmse_final_kernel(
    const float* __restrict__ partial, float* __restrict__ out) {
  const int tid = threadIdx.x;
  float loss = 0.0f, wsum = 0.0f;
  if (tid < NBC) {
    float e = 0.0f, c = 0.0f;
#pragma unroll
    for (int g = 0; g < G_PER_SLICE; ++g) {
      e += partial[2 * (tid * G_PER_SLICE + g) + 0];
      c += partial[2 * (tid * G_PER_SLICE + g) + 1];
    }
    const float w   = sqrtf(c * (1.0f / (float)SLICE_ELEMS));
    const float mse = (c > 0.0f) ? (e / c) : 0.0f;
    loss = mse * w;
    wsum = w;
  }
#pragma unroll
  for (int off = 16; off > 0; off >>= 1) {
    loss += __shfl_xor(loss, off, 32);
    wsum += __shfl_xor(wsum, off, 32);
  }
  __shared__ float sl[8], sw[8];
  if ((tid & 31) == 0) { sl[tid >> 5] = loss; sw[tid >> 5] = wsum; }
  __syncthreads();
  if (tid == 0) {
    float L = 0.0f, W = 0.0f;
#pragma unroll
    for (int i = 0; i < 8; ++i) { L += sl[i]; W += sw[i]; }
    out[0] = (W > 0.0f) ? (L / W) : 0.0f;
  }
}

extern "C" void kernel_launch(void* const* d_in, const int* in_sizes, int n_in,
                              void* d_out, int out_size, void* d_ws, size_t ws_size,
                              hipStream_t stream) {
  const float* outs = (const float*)d_in[0];
  const float* tgts = (const float*)d_in[1];
  float* partial = (float*)d_ws;   // NBC * G_PER_SLICE * 2 floats = 13.3 KB

  wmse_partial_kernel<<<NBC * G_PER_SLICE, THREADS, 0, stream>>>(outs, tgts, partial);
  wmse_final_kernel<<<1, 256, 0, stream>>>(partial, (float*)d_out);
}
